// GRN_30846455120740
// MI455X (gfx1250) — compile-verified, matrix operand order checked by reading, not run on hardware
//
#include <hip/hip_runtime.h>
#include <hip/hip_bf16.h>

typedef __attribute__((ext_vector_type(16))) _Float16 v16h;
typedef __attribute__((ext_vector_type(8)))  float    v8f;

// ---------------------------------------------------------------------------
// Elementwise / setup kernels
// ---------------------------------------------------------------------------
__global__ void zero_f32(float* p, long n) {
  long i = (long)blockIdx.x * blockDim.x + threadIdx.x;
  if (i < n) p[i] = 0.f;
}

__global__ void deg_kernel(const int* __restrict__ col, float* __restrict__ deg, int E) {
  int i = blockIdx.x * blockDim.x + threadIdx.x;
  if (i < E) unsafeAtomicAdd(&deg[col[i]], 1.0f);
}

__global__ void dinv_kernel(const float* __restrict__ deg, float* __restrict__ dinv, int n) {
  int i = blockIdx.x * blockDim.x + threadIdx.x;
  if (i < n) { float d = deg[i]; dinv[i] = (d > 0.f) ? rsqrtf(d) : 0.f; }
}

__global__ void norm_kernel(const int* __restrict__ row, const int* __restrict__ col,
                            const float* __restrict__ dinv, float* __restrict__ nrm, int E) {
  int i = blockIdx.x * blockDim.x + threadIdx.x;
  if (i < E) nrm[i] = dinv[row[i]] * dinv[col[i]];
}

// ---------------------------------------------------------------------------
// Pack fp32 matrix -> WMMA-native tiled f16 layout (ISA 7.12.2, wave32).
// Tile (rt,kt) holds 16(M/N) x 32(K) f16.  dst index = ((rt*KT+kt)*32 + lane)*16 + e
// source coords: r = rt*16 + (lane&15); c = kt*32 + 8*(lane>>4) + (e&7) + 16*(e>>3)
// Same layout serves A (rows=M) and B (pack the transposed weight: rows=N).
// ---------------------------------------------------------------------------
__global__ void pack_f16(const float* __restrict__ src, long s_row, long s_col,
                         int KT, long total, _Float16* __restrict__ dst) {
  long i = (long)blockIdx.x * blockDim.x + threadIdx.x;
  if (i >= total) return;
  long tile = i >> 9;
  int within = (int)(i & 511);
  int l = within >> 4, e = within & 15;
  long rt = tile / KT, kt = tile % KT;
  long r = rt * 16 + (l & 15);
  long c = kt * 32 + 8 * (l >> 4) + (e & 7) + 16 * (e >> 3);
  dst[i] = (_Float16)src[r * s_row + c * s_col];
}

// ---------------------------------------------------------------------------
// Batched GEMM: C[b] = A[b] @ B[b],  A packed [rowTiles x KT] tiles,
// B packed (transposed) [NT x KT] tiles, C row-major fp32 [rows x fo].
// One wave per 16-row strip; NT column tiles accumulated in registers.
// ---------------------------------------------------------------------------
template <int NT>
__global__ void gemm_wmma(const _Float16* __restrict__ Apk, long strideA,
                          const _Float16* __restrict__ Bpk, long strideB,
                          float* __restrict__ C, long strideC,
                          int rowTiles, int KT, int fo) {
  int wave = threadIdx.x >> 5;
  int lane = threadIdx.x & 31;
  int rt = blockIdx.x * (blockDim.x >> 5) + wave;
  if (rt >= rowTiles) return;               // wave-uniform: EXEC stays all-ones

  const v8f vzero = {0.f, 0.f, 0.f, 0.f, 0.f, 0.f, 0.f, 0.f};
  v8f acc[NT];
#pragma unroll
  for (int nt = 0; nt < NT; ++nt) acc[nt] = vzero;

  const _Float16* aB = Apk + (long)blockIdx.y * strideA + (long)rt * KT * 512 + lane * 16;
  const _Float16* bB = Bpk + (long)blockIdx.y * strideB + lane * 16;

  for (int kt = 0; kt < KT; ++kt) {
    v16h a = *(const v16h*)(aB + (long)kt * 512);
#pragma unroll
    for (int nt = 0; nt < NT; ++nt) {
      v16h b = *(const v16h*)(bB + ((long)nt * KT + kt) * 512);
      acc[nt] = __builtin_amdgcn_wmma_f32_16x16x32_f16(
          false, a, false, b, (short)0, acc[nt], false, false);
    }
  }

  // C/D layout: VGPR r, lanes 0-15 -> M=r, lanes 16-31 -> M=r+8; N = lane&15
  int hi = lane >> 4, nc = lane & 15;
  float* Cb = C + (long)blockIdx.y * strideC + (long)rt * 16 * fo;
#pragma unroll
  for (int nt = 0; nt < NT; ++nt)
#pragma unroll
    for (int r = 0; r < 8; ++r)
      Cb[(long)(r + 8 * hi) * fo + nt * 16 + nc] = acc[nt][r];
}

// ---------------------------------------------------------------------------
// SpMM scatter: out[b, col[e], :] += in[b, row[e], :] * norm[e]
// One thread per (edge, float4 feature chunk); native f32 atomics into L2.
// ---------------------------------------------------------------------------
__global__ void spmm_scatter(const float* __restrict__ in, long strideIn,
                             const int* __restrict__ row, const int* __restrict__ col,
                             const float* __restrict__ nrm,
                             float* __restrict__ out, long strideOut, int E, int fo) {
  int fo4 = fo >> 2;
  long idx = (long)blockIdx.x * blockDim.x + threadIdx.x;
  if (idx >= (long)E * fo4) return;
  int e = (int)(idx / fo4);
  int f = (int)(idx % fo4) * 4;
  float nv = nrm[e];
  const float* ip = in + (long)blockIdx.y * strideIn + (long)row[e] * fo + f;
  float4 v = *(const float4*)ip;
  float* op = out + (long)blockIdx.y * strideOut + (long)col[e] * fo + f;
  unsafeAtomicAdd(op + 0, v.x * nv);
  unsafeAtomicAdd(op + 1, v.y * nv);
  unsafeAtomicAdd(op + 2, v.z * nv);
  unsafeAtomicAdd(op + 3, v.w * nv);
}

// ---------------------------------------------------------------------------
// Epilogue kernels
// ---------------------------------------------------------------------------
__global__ void combine_t0(const float* __restrict__ p, const float* __restrict__ root,
                           const float* __restrict__ b, int bidx,
                           float* __restrict__ out, long n) {
  long i = (long)blockIdx.x * blockDim.x + threadIdx.x;
  if (i >= n) return;
  float v = p[i] + root[i] + b[bidx];
  out[i] = (v > 0.f) ? v : 0.2f * v;
}

__global__ void combine_t1(const float* __restrict__ p, const float* __restrict__ root,
                           const float* __restrict__ dense, const float* __restrict__ b,
                           int bidx, float* __restrict__ out, long n) {
  long i = (long)blockIdx.x * blockDim.x + threadIdx.x;
  if (i >= n) return;
  float v = p[i] + root[i] + 0.6f * dense[i] + b[bidx];
  out[i] = (v > 0.f) ? v : 0.2f * v;
}

__global__ void mean2_kernel(const float* __restrict__ a, long stride,
                             float* __restrict__ out, long n) {
  long i = (long)blockIdx.x * blockDim.x + threadIdx.x;
  if (i < n) out[i] = 0.5f * (a[i] + a[i + stride]);
}

__global__ void sum2_relu(const float* __restrict__ a, long stride,
                          float* __restrict__ out, long n) {
  long i = (long)blockIdx.x * blockDim.x + threadIdx.x;
  if (i < n) { float v = a[i] + a[i + stride]; out[i] = (v > 0.f) ? v : 0.f; }
}

// wave32 log-softmax: 32 classes == 32 lanes, one wave per node
__global__ void log_softmax32(const float* __restrict__ in, float* __restrict__ out, int N) {
  int lane = threadIdx.x & 31;
  int node = (int)(((long)blockIdx.x * blockDim.x + threadIdx.x) >> 5);
  if (node >= N) return;
  float v = in[(long)node * 32 + lane];
  float m = v;
#pragma unroll
  for (int o = 16; o > 0; o >>= 1) m = fmaxf(m, __shfl_xor(m, o, 32));
  float s = expf(v - m);
#pragma unroll
  for (int o = 16; o > 0; o >>= 1) s += __shfl_xor(s, o, 32);
  out[(long)node * 32 + lane] = v - m - logf(s);
}

// ---------------------------------------------------------------------------
static inline void launch_gemm(int NT, dim3 g, dim3 bl, hipStream_t s,
                               const _Float16* A, long sA, const _Float16* B, long sB,
                               float* C, long sC, int rowTiles, int KT, int fo) {
  if (NT == 4) gemm_wmma<4><<<g, bl, 0, s>>>(A, sA, B, sB, C, sC, rowTiles, KT, fo);
  else         gemm_wmma<2><<<g, bl, 0, s>>>(A, sA, B, sB, C, sC, rowTiles, KT, fo);
}

extern "C" void kernel_launch(void* const* d_in, const int* in_sizes, int n_in,
                              void* d_out, int out_size, void* d_ws, size_t ws_size,
                              hipStream_t stream) {
  (void)n_in; (void)out_size; (void)ws_size;
  const int F_IN = 128;
  const float* x = (const float*)d_in[0];
  const int N = in_sizes[0] / F_IN;     // 50000 (multiple of 16)
  const int E = in_sizes[1] / 2;        // 800000
  const int* row = (const int*)d_in[1];
  const int* col = row + E;

  char* ws = (char*)d_ws;
  size_t off = 0;
  auto alloc = [&](size_t bytes) -> void* {
    void* p = (void*)(ws + off);
    off += (bytes + 255) & ~(size_t)255;
    return p;
  };

  float*    deg   = (float*)alloc((size_t)N * 4);
  float*    dinv  = (float*)alloc((size_t)N * 4);
  float*    nrm   = (float*)alloc((size_t)E * 4);
  _Float16* xpk   = (_Float16*)alloc((size_t)N * 128 * 2);
  _Float16* wpk   = (_Float16*)alloc(512 * 1024);
  const size_t NF64 = (size_t)N * 64;
  float*    bufA  = (float*)alloc(2 * NF64 * 4);   // out0 / dense0 / dense1
  float*    bufB  = (float*)alloc(2 * NF64 * 4);   // propagated (scatter target)
  float*    bufC  = (float*)alloc(2 * NF64 * 4);   // root
  float*    bufD  = (float*)alloc(2 * NF64 * 4);   // out1 / out2
  float*    bufE  = (float*)alloc(2 * NF64 * 4);   // out1 @ W
  float*    dmean = (float*)alloc(NF64 * 4);
  _Float16* dmpk  = (_Float16*)alloc(NF64 * 2);
  _Float16* mpk   = (_Float16*)alloc(2 * NF64 * 2);
  float*    hbuf  = (float*)alloc(NF64 * 4);

  const int TPB = 256;
  auto nb = [](long n, int t) { return (unsigned)((n + t - 1) / t); };

  // ---- gcn_norm: deg over targets, symmetric normalization ----
  zero_f32<<<nb(N, TPB), TPB, 0, stream>>>(deg, N);
  deg_kernel<<<nb(E, TPB), TPB, 0, stream>>>(col, deg, E);
  dinv_kernel<<<nb(N, TPB), TPB, 0, stream>>>(deg, dinv, N);
  norm_kernel<<<nb(E, TPB), TPB, 0, stream>>>(row, col, dinv, nrm, E);

  struct LD { int fi, fo, K, T; };
  const LD dims[6] = {{128,64,2,2},{64,64,2,2},{64,64,2,2},{64,64,2,2},{64,64,2,2},{64,32,1,1}};

  const float* h = x;
  const int rowTilesN = N / 16;
  const dim3 gthr(8 * 32);

  for (int li = 0; li < 6; ++li) {
    const int fi = dims[li].fi, fo = dims[li].fo, K = dims[li].K, T = dims[li].T;
    const int KTi = fi / 32, KTo = fo / 32, NTo = fo / 16;
    const int base = 2 + li * 5;
    const float* Wp = (const float*)d_in[base + 0];   // (max(1,T-1), K, fo, fo)
    const float* iw = (const float*)d_in[base + 1];   // (K, fi, fo)
    const float* rw = (const float*)d_in[base + 2];   // (T, K, fi, fo)
    const float* dw = (const float*)d_in[base + 3];   // (T, K, fo, fo)
    const float* bp = (const float*)d_in[base + 4];   // (fo,)

    // pack activations (shared across all x-GEMMs of this layer)
    const long atot = (long)rowTilesN * KTi * 512;
    pack_f16<<<nb(atot, TPB), TPB, 0, stream>>>(h, fi, 1, KTi, atot, xpk);

    // pack weights (transposed -> B-operand layout)
    const long wsz = (long)fi * fo, wsz2 = (long)fo * fo;
    _Float16* iwpk  = wpk;
    _Float16* rw0pk = iwpk + K * wsz;
    _Float16* rw1pk = rw0pk + K * wsz;
    _Float16* Wpk   = rw1pk + K * wsz;
    _Float16* dwpk  = Wpk + K * wsz2;
    for (int k = 0; k < K; ++k) {
      pack_f16<<<nb(wsz, TPB), TPB, 0, stream>>>(iw + (long)k * wsz, 1, fo, KTi, wsz, iwpk + k * wsz);
      pack_f16<<<nb(wsz, TPB), TPB, 0, stream>>>(rw + (long)k * wsz, 1, fo, KTi, wsz, rw0pk + k * wsz);
      if (T > 1) {
        pack_f16<<<nb(wsz, TPB), TPB, 0, stream>>>(rw + (long)(K + k) * wsz, 1, fo, KTi, wsz, rw1pk + k * wsz);
        pack_f16<<<nb(wsz2, TPB), TPB, 0, stream>>>(Wp + (long)k * wsz2, 1, fo, KTo, wsz2, Wpk + k * wsz2);
        pack_f16<<<nb(wsz2, TPB), TPB, 0, stream>>>(dw + (long)(K + k) * wsz2, 1, fo, KTo, wsz2, dwpk + k * wsz2);
      }
    }

    const dim3 gblk(nb(rowTilesN, 8), K);
    const long NFo = (long)N * fo;
    const long knf = (long)K * NFo;
    const dim3 sblk(nb((long)E * (fo / 4), TPB), K);

    // out0 = h @ iw[k]   (also serves as dense0)
    launch_gemm(NTo, gblk, gthr, stream, xpk, 0, iwpk, wsz, bufA, NFo, rowTilesN, KTi, fo);

    // t = 0: propagate, root, leaky_relu(+b[0])
    zero_f32<<<nb(knf, TPB), TPB, 0, stream>>>(bufB, knf);
    spmm_scatter<<<sblk, TPB, 0, stream>>>(bufA, NFo, row, col, nrm, bufB, NFo, E, fo);
    launch_gemm(NTo, gblk, gthr, stream, xpk, 0, rw0pk, wsz, bufC, NFo, rowTilesN, KTi, fo);
    combine_t0<<<nb(knf, TPB), TPB, 0, stream>>>(bufB, bufC, bp, 0, bufD, knf);

    if (T > 1) {
      // m = out1 @ W[0][k]
      for (int k = 0; k < K; ++k)
        pack_f16<<<nb(NFo, TPB), TPB, 0, stream>>>(bufD + (long)k * NFo, fo, 1, KTo, NFo, mpk + (long)k * NFo);
      launch_gemm(NTo, gblk, gthr, stream, mpk, NFo, Wpk, wsz2, bufE, NFo, rowTilesN, KTo, fo);

      // propagate m
      zero_f32<<<nb(knf, TPB), TPB, 0, stream>>>(bufB, knf);
      spmm_scatter<<<sblk, TPB, 0, stream>>>(bufE, NFo, row, col, nrm, bufB, NFo, E, fo);

      // root1 = h @ rw[1][k]
      launch_gemm(NTo, gblk, gthr, stream, xpk, 0, rw1pk, wsz, bufC, NFo, rowTilesN, KTi, fo);

      // dense1 = mean_k(dense0) @ dw[1][k]
      mean2_kernel<<<nb(NFo, TPB), TPB, 0, stream>>>(bufA, NFo, dmean, NFo);
      pack_f16<<<nb(NFo, TPB), TPB, 0, stream>>>(dmean, fo, 1, KTo, NFo, dmpk);
      launch_gemm(NTo, gblk, gthr, stream, dmpk, 0, dwpk, wsz2, bufA, NFo, rowTilesN, KTo, fo);

      // out2 = leaky_relu(p + root + 0.6*dense1 + b[1])
      combine_t1<<<nb(knf, TPB), TPB, 0, stream>>>(bufB, bufC, bufA, bp, 1, bufD, knf);
    }

    if (li < 5) {
      sum2_relu<<<nb(NFo, TPB), TPB, 0, stream>>>(bufD, NFo, hbuf, NFo);  // K==2 here
      h = hbuf;
    } else {
      log_softmax32<<<nb((long)N * 32, TPB), TPB, 0, stream>>>(bufD, (float*)d_out, N);
    }
  }
}
